// TopKGOATLayer_74156905333517
// MI455X (gfx1250) — matrix operand order repaired and checked by reading, not compiled, not run on hardware
//
#include <hip/hip_runtime.h>

#define DMODEL 4096
#define NEXP   8
#define RANK   64
#define TOKENS 8192
#define KCAT   (NEXP * RANK)   // 512

typedef __attribute__((ext_vector_type(16))) __bf16       v16bf;
typedef __attribute__((ext_vector_type(8)))  float        v8f;
typedef __attribute__((ext_vector_type(4)))  unsigned int u32x4;

union Frag {
    u32x4 u[2];
    v16bf v;
};

__device__ inline unsigned short f2bf(float f) {
    unsigned int u = __builtin_bit_cast(unsigned int, f);
    u += 0x7FFFu + ((u >> 16) & 1u);           // round-to-nearest-even
    return (unsigned short)(u >> 16);
}

// ---- weight conversion: lora_A [E][R][D] f32 -> bf16, same layout ----
__global__ __launch_bounds__(256) void conv_lora_A(const float* __restrict__ A,
                                                   unsigned short* __restrict__ Abf) {
    int i = blockIdx.x * 256 + threadIdx.x;
    Abf[i] = f2bf(A[i]);
}

// ---- lora_B [E][D][R] f32 -> BT [D][E*R] bf16 (K-contiguous rows) ----
__global__ __launch_bounds__(256) void conv_lora_B(const float* __restrict__ B,
                                                   unsigned short* __restrict__ BT) {
    int i = blockIdx.x * 256 + threadIdx.x;    // over D*KCAT
    int d = i >> 9;
    int j = i & 511;
    int e = j >> 6;
    int r = j & 63;
    BT[i] = f2bf(B[((size_t)e * DMODEL + d) * RANK + r]);
}

// ---- gate (softmax top-2 renorm) + x f32 -> bf16 conversion; 1 wave/token ----
__global__ __launch_bounds__(256) void gate_convert(const float* __restrict__ x,
                                                    const float* __restrict__ gw,
                                                    unsigned short* __restrict__ xb,
                                                    float* __restrict__ wfull) {
    int wave = threadIdx.x >> 5;
    int lane = threadIdx.x & 31;
    int t    = blockIdx.x * 8 + wave;

    const float*    xr  = x  + (size_t)t * DMODEL;
    unsigned short* xbr = xb + (size_t)t * DMODEL;

    float acc[NEXP];
#pragma unroll
    for (int e = 0; e < NEXP; ++e) acc[e] = 0.f;

    for (int k = lane; k < DMODEL; k += 32) {
        float xv = xr[k];
        xbr[k] = f2bf(xv);
#pragma unroll
        for (int e = 0; e < NEXP; ++e)
            acc[e] += xv * gw[e * DMODEL + k];
    }
#pragma unroll
    for (int e = 0; e < NEXP; ++e) {
#pragma unroll
        for (int off = 16; off > 0; off >>= 1)
            acc[e] += __shfl_xor(acc[e], off, 32);
    }
    if (lane == 0) {
        // top-2 of softmax == top-2 of logits; softmax denom cancels in renorm
        float m1 = acc[0]; int i1 = 0;
#pragma unroll
        for (int e = 1; e < NEXP; ++e)
            if (acc[e] > m1) { m1 = acc[e]; i1 = e; }
        float m2 = -3.4e38f; int i2 = 0;
#pragma unroll
        for (int e = 0; e < NEXP; ++e)
            if (e != i1 && acc[e] > m2) { m2 = acc[e]; i2 = e; }
        float p2 = __expf(m2 - m1);
        float s  = 1.0f + p2;
#pragma unroll
        for (int e = 0; e < NEXP; ++e)
            wfull[t * NEXP + e] = (e == i1) ? (1.0f / s) : ((e == i2) ? (p2 / s) : 0.f);
    }
}

// ---- GEMM1: Ys[t, e*64+r] = w_full[t,e] * sum_k xb[t,k] * A_e[r,k]  (bf16 out)
// grid (Mtiles=512, E=8), 4 waves/block, one 16x16 n-tile per wave, K=4096
__global__ __launch_bounds__(128) void gemm1_scale(const unsigned short* __restrict__ xb,
                                                   const unsigned short* __restrict__ Abf,
                                                   const float* __restrict__ wfull,
                                                   unsigned short* __restrict__ Ys) {
    int mtile = blockIdx.x;
    int e     = blockIdx.y;
    int wave  = threadIdx.x >> 5;
    int lane  = threadIdx.x & 31;
    int m     = lane & 15;
    int h     = lane >> 4;
    int r     = wave * 16 + m;

    const unsigned short* arow = xb  + (size_t)(mtile * 16 + m) * DMODEL;
    const unsigned short* brow = Abf + (size_t)(e * RANK + r) * DMODEL;

    v8f acc = {};
    for (int k0 = 0; k0 < DMODEL; k0 += 32) {
        __builtin_prefetch(brow + k0 + 512, 0, 1);
        Frag a, b;
        a.u[0] = *(const u32x4*)(arow + k0 + h * 8);
        a.u[1] = *(const u32x4*)(arow + k0 + 16 + h * 8);
        b.u[0] = *(const u32x4*)(brow + k0 + h * 8);
        b.u[1] = *(const u32x4*)(brow + k0 + 16 + h * 8);
        acc = __builtin_amdgcn_wmma_f32_16x16x32_bf16(false, a.v, false, b.v,
                                                      (short)0, acc, false, false);
    }
    // D layout: lane holds col n=m, rows h*8+j -> scale per-row and store bf16
    int rowbase = mtile * 16 + h * 8;
#pragma unroll
    for (int j = 0; j < 8; ++j) {
        int   t = rowbase + j;
        float w = wfull[t * NEXP + e];
        Ys[(size_t)t * KCAT + e * RANK + wave * 16 + m] = f2bf(acc[j] * w);
    }
}

// ---- GEMM2: out[T,4096] = Ys[T,512] @ BT^T; Ys tile staged in LDS via
// GLOBAL_LOAD_ASYNC_TO_LDS (ASYNCcnt path, no VGPR bounce) ----
// grid (Mtiles=512, Ngroups=32), 8 waves/block, one 16x16 out tile per wave
__global__ __launch_bounds__(256) void gemm2(const unsigned short* __restrict__ Ys,
                                             const unsigned short* __restrict__ BT,
                                             float* __restrict__ out) {
    __shared__ unsigned short atile[16 * KCAT];   // 16 KB
    int mtile = blockIdx.x;
    int ng    = blockIdx.y;
    int wave  = threadIdx.x >> 5;
    int lane  = threadIdx.x & 31;

    // async copy of contiguous 16x512 bf16 tile into LDS: 256 lanes x 4 x 16B
    {
        const char* gsrc    = (const char*)(Ys + (size_t)mtile * 16 * KCAT);
        unsigned    ldsbase = (unsigned)(unsigned long long)(void*)atile;
#pragma unroll
        for (int i = 0; i < 4; ++i) {
            unsigned byteoff = (threadIdx.x + i * 256u) * 16u;
            asm volatile("global_load_async_to_lds_b128 %0, %1, off"
                         :: "v"(ldsbase + byteoff), "v"(gsrc + byteoff)
                         : "memory");
        }
        asm volatile("s_wait_asynccnt 0x0" ::: "memory");
    }
    __syncthreads();

    int m = lane & 15;
    int h = lane >> 4;
    int ntile = ng * 8 + wave;
    const unsigned short* brow = BT + (size_t)(ntile * 16 + m) * KCAT;
    const unsigned short* arow = atile + m * KCAT;

    v8f acc = {};
#pragma unroll
    for (int k0 = 0; k0 < KCAT; k0 += 32) {
        Frag a, b;
        a.u[0] = *(const u32x4*)(arow + k0 + h * 8);        // ds_load_b128
        a.u[1] = *(const u32x4*)(arow + k0 + 16 + h * 8);
        b.u[0] = *(const u32x4*)(brow + k0 + h * 8);
        b.u[1] = *(const u32x4*)(brow + k0 + 16 + h * 8);
        acc = __builtin_amdgcn_wmma_f32_16x16x32_bf16(false, a.v, false, b.v,
                                                      (short)0, acc, false, false);
    }
    float* orow = out + (size_t)(mtile * 16 + h * 8) * DMODEL + ntile * 16 + m;
#pragma unroll
    for (int j = 0; j < 8; ++j)
        orow[(size_t)j * DMODEL] = acc[j];
}

extern "C" void kernel_launch(void* const* d_in, const int* in_sizes, int n_in,
                              void* d_out, int out_size, void* d_ws, size_t ws_size,
                              hipStream_t stream) {
    const float* x  = (const float*)d_in[0];
    const float* gw = (const float*)d_in[1];
    const float* lA = (const float*)d_in[2];
    const float* lB = (const float*)d_in[3];
    float* out = (float*)d_out;

    char*  ws  = (char*)d_ws;
    size_t off = 0;
    unsigned short* xb    = (unsigned short*)(ws + off); off += (size_t)TOKENS * DMODEL * 2;       // 64 MB
    float*          wfull = (float*)(ws + off);          off += (size_t)TOKENS * NEXP * 4;         // 256 KB
    unsigned short* Abf   = (unsigned short*)(ws + off); off += (size_t)NEXP * RANK * DMODEL * 2;  // 4 MB
    unsigned short* BT    = (unsigned short*)(ws + off); off += (size_t)DMODEL * KCAT * 2;         // 4 MB
    unsigned short* Ys    = (unsigned short*)(ws + off);                                           // 8 MB

    conv_lora_A<<<(NEXP * RANK * DMODEL) / 256, 256, 0, stream>>>(lA, Abf);
    conv_lora_B<<<(DMODEL * KCAT) / 256, 256, 0, stream>>>(lB, BT);
    gate_convert<<<TOKENS / 8, 256, 0, stream>>>(x, gw, xb, wfull);
    gemm1_scale<<<dim3(TOKENS / 16, NEXP), 128, 0, stream>>>(xb, Abf, wfull, Ys);
    gemm2<<<dim3(TOKENS / 16, DMODEL / 128), 256, 0, stream>>>(Ys, BT, out);
}
